// QuaternionCausalSelfAttention_12756052869298
// MI455X (gfx1250) — matrix operand order repaired and checked by reading, not compiled, not run on hardware
//
#include <hip/hip_runtime.h>
#include <hip/hip_bf16.h>
#include <stdint.h>

// ---------------------------------------------------------------------------
// QuaternionCausalSelfAttention for MI455X (gfx1250), wave32 + WMMA bf16.
// All matmuls on v_wmma_f32_16x16x32_bf16 (f32 accumulate). Both GEMM and
// attention stage LDS tiles with the Tensor Data Mover (tensor_load_to_lds,
// TENSORcnt), double-buffered: tile s+1 DMAs while tile s feeds WMMA.
// Weights are pre-transposed to bf16 [N][K] so every WMMA operand (A and B)
// is a contiguous K-run in LDS -> ds_load_b128-only fragment feeds.
// ---------------------------------------------------------------------------

typedef __attribute__((ext_vector_type(16))) __bf16          v16bf;
typedef __attribute__((ext_vector_type(8)))  float           v8f;
typedef __attribute__((ext_vector_type(16))) unsigned short  v16u;
typedef __attribute__((ext_vector_type(4)))  unsigned        v4u;
typedef __attribute__((ext_vector_type(8)))  int             v8i_;
typedef __attribute__((ext_vector_type(4)))  int             v4i_;

#define DEVINL __device__ __forceinline__

static constexpr int B_  = 4;
static constexpr int T_  = 2048;
static constexpr int C_  = 2048;
static constexpr int H_  = 16;
static constexpr int HD_ = 128;
static constexpr int BT_ = B_ * T_;
static constexpr float QSCALE = 0.08838834764831845f;  // 1/sqrt(128)

#if __has_builtin(__builtin_amdgcn_tensor_load_to_lds) && \
    __has_builtin(__builtin_amdgcn_s_wait_tensorcnt)
#define USE_TDM 1
#else
#define USE_TDM 0
#endif

// fp32 -> bf16 round-to-nearest-even.
DEVINL unsigned short f2bf(float f) {
  unsigned u = __float_as_uint(f);
  unsigned r = u + 0x7fffu + ((u >> 16) & 1u);
  return (unsigned short)(r >> 16);
}

// --- WMMA fragment loaders (contiguous k = 16*(lane/16) + s for A and B) -----------
DEVINL v16bf load_a_frag(const unsigned short* tile, int rowBase, int ld,
                         int k0, int lane) {
  const unsigned short* p = tile + (rowBase + (lane & 15)) * ld + k0;
  const int half8 = (lane >> 4) << 3;
  v16u r;
#pragma unroll
  for (int j = 0; j < 8; ++j) {
    int kk = 2 * half8 + 2 * j;
    unsigned v = *(const unsigned*)(p + kk);
    r[2 * j]     = (unsigned short)v;
    r[2 * j + 1] = (unsigned short)(v >> 16);
  }
  return __builtin_bit_cast(v16bf, r);
}

DEVINL v16bf load_b_frag(const unsigned short* tile, int nBase, int ld,
                         int k0, int lane) {
  const unsigned short* p =
      tile + (nBase + (lane & 15)) * ld + k0 + ((lane >> 4) << 4);
  v16u r;
#pragma unroll
  for (int j = 0; j < 8; ++j) {
    unsigned v = *(const unsigned*)(p + 2 * j);
    r[2 * j]     = (unsigned short)v;
    r[2 * j + 1] = (unsigned short)(v >> 16);
  }
  return __builtin_bit_cast(v16bf, r);
}

DEVINL v8f wmma_bf16(v16bf a, v16bf b, v8f c) {
  return __builtin_amdgcn_wmma_f32_16x16x32_bf16(false, a, false, b,
                                                 (short)0, c, false, false);
}

DEVINL v8f v8f_zero() {
  v8f z = {0.f, 0.f, 0.f, 0.f, 0.f, 0.f, 0.f, 0.f};
  return z;
}

#if USE_TDM
// 2-D tile DMA global->LDS via Tensor Data Mover (D# per ISA §8.3/8.4).
// tensorW: row pitch (elements, contiguous dim); tensorH: #rows;
// tileW x tileH tile. data_size = 2 bytes. gsrc points at the tile start.
DEVINL void tdm_load_2d(unsigned ldsOff, const void* gsrc, unsigned tensorW,
                        unsigned tensorH, unsigned tileW, unsigned tileH) {
  unsigned long long ga = (unsigned long long)(size_t)gsrc;
  v4u g0 = {1u,                                     // count=1, user mode
            ldsOff,                                 // lds_addr (bytes)
            (unsigned)ga,                           // global_addr[31:0]
            ((unsigned)(ga >> 32) & 0x01FFFFFFu) | 0x80000000u};  // type=2
  v8i_ g1;
  g1[0] = (int)(1u << 16);                          // data_size = 2B
  g1[1] = (int)((tensorW & 0xFFFFu) << 16);         // tensor_dim0[15:0]
  g1[2] = (int)((tensorW >> 16) | ((tensorH & 0xFFFFu) << 16));
  g1[3] = (int)((tensorH >> 16) | (tileW << 16));   // tile_dim0
  g1[4] = (int)tileH;                               // tile_dim1 (tile_dim2=0)
  g1[5] = (int)tensorW;                             // tensor_dim0_stride lo
  g1[6] = 0;
  g1[7] = 0;
  v4i_ z4 = {0, 0, 0, 0};
#if __clang_major__ >= 23
  v8i_ z8 = {0, 0, 0, 0, 0, 0, 0, 0};
  __builtin_amdgcn_tensor_load_to_lds(g0, g1, z4, z4, z8, 0);
#else
  __builtin_amdgcn_tensor_load_to_lds(g0, g1, z4, z4, 0);
#endif
}
#endif

// ---------------------------------------------------------------------------
// Kernel 0: fp32 -> bf16 bulk convert (activations).
// ---------------------------------------------------------------------------
__global__ __launch_bounds__(256) void k_f32_to_bf16(
    const float* __restrict__ src, unsigned short* __restrict__ dst, int n4) {
  int i = blockIdx.x * 256 + threadIdx.x;
  if (i < n4) {
    float4 v = ((const float4*)src)[i];
    unsigned u0 = (unsigned)f2bf(v.x) | ((unsigned)f2bf(v.y) << 16);
    unsigned u1 = (unsigned)f2bf(v.z) | ((unsigned)f2bf(v.w) << 16);
    ((uint2*)dst)[i] = make_uint2(u0, u1);
  }
}

// ---------------------------------------------------------------------------
// Kernel 1: weight prep: W f32 [K][N] -> Wt bf16 [N][K] (LDS tile transpose,
// coalesced on both sides; 2-byte scatter confined to LDS). Runs once per
// weight, amortized over all 64 M-supertiles of the GEMMs.
// ---------------------------------------------------------------------------
__global__ __launch_bounds__(128) void k_transpose_w(
    const float* __restrict__ W, unsigned short* __restrict__ Wt,
    int K, int N) {
  __shared__ unsigned short Ls[128 * 64];  // [n][k]
  const int tid   = threadIdx.x;
  const int kBase = blockIdx.x * 64;
  const int nBase = blockIdx.y * 128;
  for (int i = tid; i < 64 * 32; i += 128) {
    int row = i >> 5;          // k-local
    int c4  = (i & 31) << 2;   // n-local
    float4 v = *(const float4*)(W + (size_t)(kBase + row) * N + nBase + c4);
    Ls[(c4 + 0) * 64 + row] = f2bf(v.x);
    Ls[(c4 + 1) * 64 + row] = f2bf(v.y);
    Ls[(c4 + 2) * 64 + row] = f2bf(v.z);
    Ls[(c4 + 3) * 64 + row] = f2bf(v.w);
  }
  __syncthreads();
  for (int i = tid; i < 1024; i += 128) {
    int n = i >> 3;
    int q = (i & 7) << 3;
    *(uint4*)(Wt + (size_t)(nBase + n) * K + kBase + q) =
        *(const uint4*)(Ls + n * 64 + q);
  }
}

// ---------------------------------------------------------------------------
// Kernel 2: C[M,N] f32 = A[M,K]bf16 @ Bt[N,K]bf16 (B pre-transposed).
// Block tile 128x128, KTILE 64, 8 waves x (2x4) WMMA accumulators.
// A and B tiles are identical 128x64 K-major slabs: both DMA'd by TDM,
// double-buffered on TENSORcnt so stage s+1 streams during stage s compute.
// ---------------------------------------------------------------------------
__global__ __launch_bounds__(256) void k_gemm_bf16(
    const unsigned short* __restrict__ A, const unsigned short* __restrict__ Bt,
    float* __restrict__ C, int M, int N, int K) {
  __shared__ unsigned short As[2][128 * 64];   // [row][k]
  __shared__ unsigned short Bs[2][128 * 64];   // [n][k]
  const int tid   = threadIdx.x;
  const int lane  = tid & 31;
  const int wid   = tid >> 5;
  const int wm    = (wid & 3) << 5;
  const int wn    = (wid >> 2) << 6;
  const int tileM = blockIdx.y * 128;
  const int tileN = blockIdx.x * 128;
  const unsigned short* Arow = A  + (size_t)tileM * K;
  const unsigned short* Brow = Bt + (size_t)tileN * K;
  const int nStages = K >> 6;

  v8f acc[2][4];
#pragma unroll
  for (int i = 0; i < 2; ++i)
#pragma unroll
    for (int j = 0; j < 4; ++j) acc[i][j] = v8f_zero();

#if USE_TDM
  if (tid < 32) {
    tdm_load_2d((unsigned)(size_t)(void*)As[0], Arow, K, 128, 64, 128);
    tdm_load_2d((unsigned)(size_t)(void*)Bs[0], Brow, K, 128, 64, 128);
  }
#endif

  for (int s = 0; s < nStages; ++s) {
    const int buf = s & 1;
#if USE_TDM
    if (tid < 32) __builtin_amdgcn_s_wait_tensorcnt(0);
    __syncthreads();  // stage s visible; buf^1 free (stage s-1 consumed)
    if (s + 1 < nStages && tid < 32) {
      tdm_load_2d((unsigned)(size_t)(void*)As[buf ^ 1], Arow + (s + 1) * 64,
                  K, 128, 64, 128);
      tdm_load_2d((unsigned)(size_t)(void*)Bs[buf ^ 1], Brow + (s + 1) * 64,
                  K, 128, 64, 128);
    }
#else
    const int k0 = s << 6;
    __syncthreads();
#pragma unroll
    for (int it = 0; it < 4; ++it) {
      int idx = tid + it * 256;
      int row = idx >> 3;
      int q   = (idx & 7) << 3;
      *(uint4*)(As[buf] + row * 64 + q) =
          *(const uint4*)(Arow + (size_t)row * K + k0 + q);
      *(uint4*)(Bs[buf] + row * 64 + q) =
          *(const uint4*)(Brow + (size_t)row * K + k0 + q);
    }
    if (k0 + 64 < K) {
      __builtin_prefetch(Arow + (size_t)(tid >> 1) * K + k0 + 64, 0, 1);
      __builtin_prefetch(Brow + (size_t)(tid >> 1) * K + k0 + 64, 0, 1);
    }
    __syncthreads();
#endif
#pragma unroll
    for (int kc = 0; kc < 64; kc += 32) {
      v16bf a0 = load_a_frag(As[buf], wm + 0,  64, kc, lane);
      v16bf a1 = load_a_frag(As[buf], wm + 16, 64, kc, lane);
#pragma unroll
      for (int fn = 0; fn < 4; ++fn) {
        v16bf b = load_b_frag(Bs[buf], wn + fn * 16, 64, kc, lane);
        acc[0][fn] = wmma_bf16(a0, b, acc[0][fn]);
        acc[1][fn] = wmma_bf16(a1, b, acc[1][fn]);
      }
    }
  }
  const int rlo  = (lane >> 4) << 3;
  const int col0 = lane & 15;
#pragma unroll
  for (int fm = 0; fm < 2; ++fm)
#pragma unroll
    for (int fn = 0; fn < 4; ++fn)
#pragma unroll
      for (int r = 0; r < 8; ++r) {
        int row = tileM + wm + fm * 16 + r + rlo;
        int col = tileN + wn + fn * 16 + col0;
        C[(size_t)row * N + col] = acc[fm][fn][r];
      }
}

// ---------------------------------------------------------------------------
// Kernel 3: fused RoPE + RMS (q,k), qnormalize(k), t = qmul(conj(kn), v).
// One wave per (b,t,h) row; lane = one quaternion.
// ---------------------------------------------------------------------------
__global__ __launch_bounds__(256) void k_rope_rms(
    const float* __restrict__ qr, const float* __restrict__ kr,
    const float* __restrict__ vr, const float* __restrict__ cosv,
    const float* __restrict__ sinv, float* __restrict__ q_t,
    unsigned short* __restrict__ q_b, unsigned short* __restrict__ k_b,
    float* __restrict__ t_t) {
  const int lane = threadIdx.x & 31;
  const int gw   = blockIdx.x * 8 + (threadIdx.x >> 5);
  const int h    = gw & (H_ - 1);
  const int bt   = gw >> 4;
  const int t    = bt & (T_ - 1);
  const int b    = bt >> 11;
  const size_t inOff  = (size_t)bt * C_ + h * HD_ + lane * 4;
  const size_t outOff = ((size_t)(b * H_ + h) * T_ + t) * HD_ + lane * 4;

  float4 q = *(const float4*)(qr + inOff);
  float4 k = *(const float4*)(kr + inOff);
  float4 v = *(const float4*)(vr + inOff);

  const int ci = (lane & 15) * 4;
  float4 cs = *(const float4*)(cosv + (size_t)t * 64 + ci);
  float4 sn = *(const float4*)(sinv + (size_t)t * 64 + ci);
  float4 q2, k2;
  q2.x = __shfl_xor(q.x, 16, 32); q2.y = __shfl_xor(q.y, 16, 32);
  q2.z = __shfl_xor(q.z, 16, 32); q2.w = __shfl_xor(q.w, 16, 32);
  k2.x = __shfl_xor(k.x, 16, 32); k2.y = __shfl_xor(k.y, 16, 32);
  k2.z = __shfl_xor(k.z, 16, 32); k2.w = __shfl_xor(k.w, 16, 32);
  const bool hi = lane >= 16;
  float4 qy, ky;
  if (hi) {  // y2 = -x1*sin + x2*cos, x1 = partner (first half)
    qy.x = -q2.x * sn.x + q.x * cs.x; qy.y = -q2.y * sn.y + q.y * cs.y;
    qy.z = -q2.z * sn.z + q.z * cs.z; qy.w = -q2.w * sn.w + q.w * cs.w;
    ky.x = -k2.x * sn.x + k.x * cs.x; ky.y = -k2.y * sn.y + k.y * cs.y;
    ky.z = -k2.z * sn.z + k.z * cs.z; ky.w = -k2.w * sn.w + k.w * cs.w;
  } else {   // y1 = x1*cos + x2*sin, x2 = partner (second half)
    qy.x = q.x * cs.x + q2.x * sn.x; qy.y = q.y * cs.y + q2.y * sn.y;
    qy.z = q.z * cs.z + q2.z * sn.z; qy.w = q.w * cs.w + q2.w * sn.w;
    ky.x = k.x * cs.x + k2.x * sn.x; ky.y = k.y * cs.y + k2.y * sn.y;
    ky.z = k.z * cs.z + k2.z * sn.z; ky.w = k.w * cs.w + k2.w * sn.w;
  }

  float sq = qy.x * qy.x + qy.y * qy.y + qy.z * qy.z + qy.w * qy.w;
  float sk = ky.x * ky.x + ky.y * ky.y + ky.z * ky.z + ky.w * ky.w;
#pragma unroll
  for (int mm = 1; mm <= 16; mm <<= 1) {
    sq += __shfl_xor(sq, mm, 32);
    sk += __shfl_xor(sk, mm, 32);
  }
  const float rq = rsqrtf(sq * (1.0f / HD_) + 1e-6f);
  const float rk = rsqrtf(sk * (1.0f / HD_) + 1e-6f);
  qy.x *= rq; qy.y *= rq; qy.z *= rq; qy.w *= rq;
  ky.x *= rk; ky.y *= rk; ky.z *= rk; ky.w *= rk;

  *(float4*)(q_t + outOff) = qy;
  {
    unsigned u0 = (unsigned)f2bf(qy.x * QSCALE) |
                  ((unsigned)f2bf(qy.y * QSCALE) << 16);
    unsigned u1 = (unsigned)f2bf(qy.z * QSCALE) |
                  ((unsigned)f2bf(qy.w * QSCALE) << 16);
    *(uint2*)(q_b + outOff) = make_uint2(u0, u1);
    unsigned v0 = (unsigned)f2bf(ky.x) | ((unsigned)f2bf(ky.y) << 16);
    unsigned v1 = (unsigned)f2bf(ky.z) | ((unsigned)f2bf(ky.w) << 16);
    *(uint2*)(k_b + outOff) = make_uint2(v0, v1);
  }

  float nrm = sqrtf(ky.x * ky.x + ky.y * ky.y + ky.z * ky.z + ky.w * ky.w);
  float inn = 1.0f / fmaxf(nrm, 1e-12f);
  float aw =  ky.x * inn, ax = -ky.y * inn, ay = -ky.z * inn, az = -ky.w * inn;
  float bw = v.x, bx = v.y, by = v.z, bz = v.w;
  float4 o;
  o.x = aw * bw - ax * bx - ay * by - az * bz;
  o.y = aw * bx + ax * bw + ay * bz - az * by;
  o.z = aw * by - ax * bz + ay * bw + az * bx;
  o.w = aw * bz + ax * by - ay * bx + az * bw;
  *(float4*)(t_t + outOff) = o;
}

// ---------------------------------------------------------------------------
// Kernel 4: t_t f32 [bh][t][d] -> t_b bf16 [bh][d][t] via LDS tile transpose.
// ---------------------------------------------------------------------------
__global__ __launch_bounds__(128) void k_transpose_t(
    const float* __restrict__ t_t, unsigned short* __restrict__ t_b) {
  __shared__ unsigned short Ls[128 * 64];  // [d][t-local]
  const int tid   = threadIdx.x;
  const int bh    = blockIdx.y;
  const int tBase = blockIdx.x * 64;
  const size_t inBase = (size_t)bh * T_ * HD_;
  for (int i = tid; i < 64 * 32; i += 128) {
    int row = i >> 5;
    int c4  = (i & 31) << 2;
    float4 v = *(const float4*)(t_t + inBase + (size_t)(tBase + row) * HD_ + c4);
    Ls[(c4 + 0) * 64 + row] = f2bf(v.x);
    Ls[(c4 + 1) * 64 + row] = f2bf(v.y);
    Ls[(c4 + 2) * 64 + row] = f2bf(v.z);
    Ls[(c4 + 3) * 64 + row] = f2bf(v.w);
  }
  __syncthreads();
  unsigned short* out = t_b + (size_t)bh * HD_ * T_ + tBase;
  for (int i = tid; i < 1024; i += 128) {
    int d = i >> 3;
    int q = (i & 7) << 3;
    *(uint4*)(out + (size_t)d * T_ + q) = *(const uint4*)(Ls + d * 64 + q);
  }
}

// ---------------------------------------------------------------------------
// Kernel 5: causal flash attention. grid=(T/64, B*H), block=128 (4 waves),
// wave w owns 16 query rows. K/T tiles double-buffered via TDM; next tile's
// DMA overlaps current tile's WMMAs (s_wait_tensorcnt + workgroup barrier).
// ---------------------------------------------------------------------------
__global__ __launch_bounds__(128) void k_attn(
    const unsigned short* __restrict__ qb, const unsigned short* __restrict__ kb,
    const unsigned short* __restrict__ tb, float* __restrict__ agg) {
  __shared__ unsigned short Qs[64 * 128];       // [qrow][d] (pre-scaled bf16)
  __shared__ unsigned short Ks[2][64 * 128];    // [key][d]
  __shared__ unsigned short Ts[2][128 * 64];    // [d][key]
  __shared__ unsigned short Ps[4][16 * 64];     // wave-private P tiles
  const int tid   = threadIdx.x;
  const int lane  = tid & 31;
  const int w     = tid >> 5;
  const int qt    = blockIdx.x;
  const int bh    = blockIdx.y;
  const int qBase = qt * 64;
  const unsigned short* qbh = qb + (size_t)bh * T_ * HD_;
  const unsigned short* kbh = kb + (size_t)bh * T_ * HD_;
  const unsigned short* tbh = tb + (size_t)bh * HD_ * T_;

#if USE_TDM
  if (tid < 32) {
    tdm_load_2d((unsigned)(size_t)(void*)Qs, qbh + (size_t)qBase * HD_,
                HD_, T_, HD_, 64);
    tdm_load_2d((unsigned)(size_t)(void*)Ks[0], kbh, HD_, T_, HD_, 64);
    tdm_load_2d((unsigned)(size_t)(void*)Ts[0], tbh, T_, HD_, 64, HD_);
  }
#else
  for (int i = tid; i < 1024; i += 128) {
    int row = i >> 3, q = (i & 7) << 3;
    *(uint4*)(Qs + row * 128 + q) =
        *(const uint4*)(qbh + (size_t)(qBase + row) * HD_ + q);
  }
#endif

  v8f O[8];
  float m[8], l[8];
#pragma unroll
  for (int i = 0; i < 8; ++i) {
    O[i] = v8f_zero(); m[i] = -__builtin_inff(); l[i] = 0.f;
  }

  for (int kt = 0; kt <= qt; ++kt) {
    const int kBase = kt * 64;
    const int buf   = kt & 1;
#if USE_TDM
    if (tid < 32) __builtin_amdgcn_s_wait_tensorcnt(0);
    __syncthreads();  // tile kt visible to all; buf^1 free for reuse
    if (kt < qt && tid < 32) {
      tdm_load_2d((unsigned)(size_t)(void*)Ks[buf ^ 1],
                  kbh + (size_t)(kBase + 64) * HD_, HD_, T_, HD_, 64);
      tdm_load_2d((unsigned)(size_t)(void*)Ts[buf ^ 1],
                  tbh + (kBase + 64), T_, HD_, 64, HD_);
    }
#else
    __syncthreads();
    for (int i = tid; i < 1024; i += 128) {
      int row = i >> 3, q = (i & 7) << 3;
      *(uint4*)(Ks[buf] + row * 128 + q) =
          *(const uint4*)(kbh + (size_t)(kBase + row) * HD_ + q);
    }
    for (int i = tid; i < 1024; i += 128) {
      int d = i >> 3, q = (i & 7) << 3;
      *(uint4*)(Ts[buf] + d * 64 + q) =
          *(const uint4*)(tbh + (size_t)d * T_ + kBase + q);
    }
    __syncthreads();
#endif

    // ---- S = Q @ K^T
    v8f S[4];
#pragma unroll
    for (int fn = 0; fn < 4; ++fn) S[fn] = v8f_zero();
#pragma unroll
    for (int ks = 0; ks < 4; ++ks) {
      v16bf a = load_a_frag(Qs, w * 16, 128, ks * 32, lane);
#pragma unroll
      for (int fn = 0; fn < 4; ++fn) {
        v16bf b = load_b_frag(Ks[buf], fn * 16, 128, ks * 32, lane);
        S[fn] = wmma_bf16(a, b, S[fn]);
      }
    }

    // ---- causal mask (diagonal tile only)
    if (kt == qt) {
      const int keyc = kBase + (lane & 15);
      const int rowq = qBase + w * 16 + ((lane >> 4) << 3);
#pragma unroll
      for (int fn = 0; fn < 4; ++fn)
#pragma unroll
        for (int r = 0; r < 8; ++r)
          if (keyc + fn * 16 > rowq + r) S[fn][r] = -__builtin_inff();
    }

    // ---- online softmax (xor-shuffle row reductions, 16-lane groups)
    float alpha[8];
#pragma unroll
    for (int r = 0; r < 8; ++r) {
      float mx = fmaxf(fmaxf(S[0][r], S[1][r]), fmaxf(S[2][r], S[3][r]));
      mx = fmaxf(mx, __shfl_xor(mx, 1, 32));
      mx = fmaxf(mx, __shfl_xor(mx, 2, 32));
      mx = fmaxf(mx, __shfl_xor(mx, 4, 32));
      mx = fmaxf(mx, __shfl_xor(mx, 8, 32));
      float mnew = fmaxf(m[r], mx);
      alpha[r] = __expf(m[r] - mnew);
      m[r] = mnew;
    }
    unsigned short* Pw = Ps[w];
#pragma unroll
    for (int r = 0; r < 8; ++r) {
      float rs = 0.f;
      const int prow = (r + ((lane >> 4) << 3)) * 64 + (lane & 15);
#pragma unroll
      for (int fn = 0; fn < 4; ++fn) {
        float p = __expf(S[fn][r] - m[r]);
        rs += p;
        Pw[prow + fn * 16] = f2bf(p);
      }
      rs += __shfl_xor(rs, 1, 32);
      rs += __shfl_xor(rs, 2, 32);
      rs += __shfl_xor(rs, 4, 32);
      rs += __shfl_xor(rs, 8, 32);
      l[r] = l[r] * alpha[r] + rs;
    }

    // ---- O = O*alpha + P @ Ttile
#pragma unroll
    for (int fn = 0; fn < 8; ++fn)
#pragma unroll
      for (int r = 0; r < 8; ++r) O[fn][r] *= alpha[r];
#pragma unroll
    for (int ks = 0; ks < 2; ++ks) {
      v16bf a = load_a_frag(Pw, 0, 64, ks * 32, lane);
#pragma unroll
      for (int fn = 0; fn < 8; ++fn) {
        v16bf b = load_b_frag(Ts[buf], fn * 16, 64, ks * 32, lane);
        O[fn] = wmma_bf16(a, b, O[fn]);
      }
    }
  }

  const int rlo  = (lane >> 4) << 3;
  const int col0 = lane & 15;
  const size_t rowBase = (size_t)bh * T_ * HD_;
#pragma unroll
  for (int r = 0; r < 8; ++r) {
    float inv = 1.0f / l[r];
    int row = qBase + w * 16 + r + rlo;
#pragma unroll
    for (int fn = 0; fn < 8; ++fn)
      agg[rowBase + (size_t)row * HD_ + fn * 16 + col0] = O[fn][r] * inv;
  }
}

// ---------------------------------------------------------------------------
// Kernel 6: y = qmul(q, agg), repack [bh][t][d] -> bf16 [b*T+t][h*128+d].
// ---------------------------------------------------------------------------
__global__ __launch_bounds__(256) void k_qmul_out(
    const float* __restrict__ q_t, const float* __restrict__ agg,
    unsigned short* __restrict__ yb) {
  const int lane = threadIdx.x & 31;
  const int gw   = blockIdx.x * 8 + (threadIdx.x >> 5);
  const int h    = gw & (H_ - 1);
  const int bt   = gw >> 4;
  const int t    = bt & (T_ - 1);
  const int b    = bt >> 11;
  const size_t tOff = ((size_t)(b * H_ + h) * T_ + t) * HD_ + lane * 4;
  float4 q = *(const float4*)(q_t + tOff);
  float4 g = *(const float4*)(agg + tOff);
  float aw = q.x, ax = q.y, ay = q.z, az = q.w;
  float bw = g.x, bx = g.y, by = g.z, bz = g.w;
  float ow = aw * bw - ax * bx - ay * by - az * bz;
  float ox = aw * bx + ax * bw + ay * bz - az * by;
  float oy = aw * by - ax * bz + ay * bw + az * bx;
  float oz = aw * bz + ax * by - ay * bx + az * bw;
  unsigned u0 = (unsigned)f2bf(ow) | ((unsigned)f2bf(ox) << 16);
  unsigned u1 = (unsigned)f2bf(oy) | ((unsigned)f2bf(oz) << 16);
  const size_t oOff = (size_t)bt * C_ + h * HD_ + lane * 4;
  *(uint2*)(yb + oOff) = make_uint2(u0, u1);
}

// ---------------------------------------------------------------------------
// Host launcher. Workspace layout (bytes), ~448 MB:
//   [  0, 32M)  xb bf16 [M][K]  -> reused as q_b after projections
//   [ 32, 64M)  Wq/Wk/Wv/Wo transposed bf16 [N][K], 8M each
//   [ 64,128M)  q_raw f32       -> reused as agg
//   [128,192M)  k_raw f32       -> reused as yb
//   [192,256M)  v_raw f32
//   [256,320M)  q_t  f32 [bh][t][d]
//   [320,384M)  t_t  f32 [bh][t][d]
//   [384,416M)  k_b  bf16 [bh][t][d]
//   [416,448M)  t_b  bf16 [bh][d][t]
// ---------------------------------------------------------------------------
extern "C" void kernel_launch(void* const* d_in, const int* in_sizes, int n_in,
                              void* d_out, int out_size, void* d_ws,
                              size_t ws_size, hipStream_t stream) {
  const float* x    = (const float*)d_in[0];
  const float* cosv = (const float*)d_in[1];
  const float* sinv = (const float*)d_in[2];
  const float* Wq   = (const float*)d_in[3];
  const float* Wk   = (const float*)d_in[4];
  const float* Wv   = (const float*)d_in[5];
  const float* Wo   = (const float*)d_in[6];

  char* ws = (char*)d_ws;
  const size_t MB = 1ull << 20;
  unsigned short* xb  = (unsigned short*)(ws);
  unsigned short* Wqt = (unsigned short*)(ws + 32 * MB);
  unsigned short* Wkt = (unsigned short*)(ws + 40 * MB);
  unsigned short* Wvt = (unsigned short*)(ws + 48 * MB);
  unsigned short* Wot = (unsigned short*)(ws + 56 * MB);
  float* q_raw = (float*)(ws + 64 * MB);
  float* k_raw = (float*)(ws + 128 * MB);
  float* v_raw = (float*)(ws + 192 * MB);
  float* q_t   = (float*)(ws + 256 * MB);
  float* t_t   = (float*)(ws + 320 * MB);
  unsigned short* k_b = (unsigned short*)(ws + 384 * MB);
  unsigned short* t_b = (unsigned short*)(ws + 416 * MB);
  unsigned short* q_b = xb;                    // reuse (xb dead after GEMMs)
  float* agg = q_raw;                          // reuse
  unsigned short* yb = (unsigned short*)k_raw; // reuse

  const int n4x = BT_ * C_ / 4;
  k_f32_to_bf16<<<n4x / 256, 256, 0, stream>>>(x, xb, n4x);
  dim3 wtg(C_ / 64, C_ / 128);
  k_transpose_w<<<wtg, 128, 0, stream>>>(Wq, Wqt, C_, C_);
  k_transpose_w<<<wtg, 128, 0, stream>>>(Wk, Wkt, C_, C_);
  k_transpose_w<<<wtg, 128, 0, stream>>>(Wv, Wvt, C_, C_);
  k_transpose_w<<<wtg, 128, 0, stream>>>(Wo, Wot, C_, C_);

  dim3 gg(C_ / 128, BT_ / 128);
  k_gemm_bf16<<<gg, 256, 0, stream>>>(xb, Wqt, q_raw, BT_, C_, C_);
  k_gemm_bf16<<<gg, 256, 0, stream>>>(xb, Wkt, k_raw, BT_, C_, C_);
  k_gemm_bf16<<<gg, 256, 0, stream>>>(xb, Wvt, v_raw, BT_, C_, C_);

  k_rope_rms<<<BT_ * H_ / 8, 256, 0, stream>>>(q_raw, k_raw, v_raw, cosv, sinv,
                                               q_t, q_b, k_b, t_t);

  dim3 tg(T_ / 64, B_ * H_);
  k_transpose_t<<<tg, 128, 0, stream>>>(t_t, t_b);

  dim3 ag(T_ / 64, B_ * H_);
  k_attn<<<ag, 128, 0, stream>>>(q_b, k_b, t_b, agg);

  k_qmul_out<<<BT_ * H_ / 8, 256, 0, stream>>>(q_t, agg, yb);

  k_gemm_bf16<<<gg, 256, 0, stream>>>(yb, Wot, (float*)d_out, BT_, C_, C_);
}
